// CausalSelfAttention_14740327759972
// MI455X (gfx1250) — matrix operand order repaired and checked by reading, not compiled, not run on hardware
//
#include <hip/hip_runtime.h>

// ---------------------------------------------------------------------------
// Causal multi-head self-attention for MI455X (gfx1250), wave32 + WMMA bf16.
//   B=4, T=2048, C=1024, H=16, hs=64.
// Pipeline: cvt(x,Wa,Wp)->bf16 ; qkv = x@Wa (WMMA GEMM) ; flash-attn (WMMA,
//           block-cooperative K/V staging in LDS) ; out = y@Wp (WMMA GEMM).
// Workspace usage: ~92 MB (all sub-buffers 16B aligned).
// ---------------------------------------------------------------------------

typedef __attribute__((ext_vector_type(16))) __bf16        v16bf;
typedef __attribute__((ext_vector_type(8)))  float         v8f;
typedef __attribute__((ext_vector_type(8)))  unsigned int  v8u;

__device__ __forceinline__ unsigned short f2bf(float f) {
    unsigned u = __builtin_bit_cast(unsigned, f);
    u += 0x7FFFu + ((u >> 16) & 1u);          // round-to-nearest-even
    return (unsigned short)(u >> 16);
}
__device__ __forceinline__ unsigned packbf(float lo, float hi) {
    return (unsigned)f2bf(lo) | ((unsigned)f2bf(hi) << 16);
}
__device__ __forceinline__ v16bf asbf(v8u u) { return __builtin_bit_cast(v16bf, u); }

// ---------------------------------------------------------------------------
// fp32 -> bf16 elementwise convert
// ---------------------------------------------------------------------------
__global__ void cvt_f32_bf16(const float* __restrict__ in,
                             unsigned short* __restrict__ out, int n) {
    for (int i = blockIdx.x * blockDim.x + threadIdx.x; i < n;
         i += gridDim.x * blockDim.x)
        out[i] = f2bf(in[i]);
}

// ---------------------------------------------------------------------------
// LDS-tiled WMMA GEMM: C[M,N] = A[M,K](bf16) * B[K,N](bf16)
// Block tile 128x128, BK=32. 256 threads = 8 waves, wave tile 32x64 (2x4 WMMA).
// LDS rows padded to 20 dwords (80B) so all fragment reads are ds_load_b128.
// ---------------------------------------------------------------------------
template <bool BF16OUT>
__global__ __launch_bounds__(256) void wmma_gemm_bf16(
    const unsigned short* __restrict__ A,
    const unsigned short* __restrict__ B,
    unsigned short* __restrict__ Cb,
    float* __restrict__ Cf,
    int M, int N, int K)
{
    __shared__ __align__(16) unsigned As[128][20];  // [m][k/2] pairs
    __shared__ __align__(16) unsigned Bt[128][20];  // [n][k-pair] transposed

    const unsigned* A32 = (const unsigned*)A;   // [M][K/2]
    const unsigned* B32 = (const unsigned*)B;   // [K][N/2]
    const int K2 = K >> 1, N2 = N >> 1;

    const int t    = threadIdx.x;
    const int m0   = blockIdx.y * 128;
    const int n0   = blockIdx.x * 128;
    const int wid  = t >> 5;
    const int lane = t & 31;
    const int lm   = lane & 15;
    const bool hiL = lane >= 16;
    const int wm   = (wid & 3) * 32;
    const int wn   = (wid >> 2) * 64;

    v8f acc[2][4] = {};

    for (int k0 = 0; k0 < K; k0 += 32) {
        __syncthreads();
        // ---- stage A tile: 128 rows x 16 dwords, two b128 loads per thread
        {
            int r  = t >> 1;
            int cg = (t & 1) * 2;
            const uint4* src = (const uint4*)&A32[(long)(m0 + r) * K2 + (k0 >> 1)];
            uint4* dst = (uint4*)&As[r][0];
            dst[cg]     = src[cg];
            dst[cg + 1] = src[cg + 1];
        }
        // ---- stage B tile transposed: coalesced u32 -> repack k-pairs -> LDS
        #pragma unroll
        for (int i = 0; i < 4; ++i) {
            int flat = i * 256 + t;
            int n2 = flat & 63;
            int kp = flat >> 6;
            int k  = 2 * kp;
            unsigned a = B32[(long)(k0 + k)     * N2 + (n0 >> 1) + n2];
            unsigned b = B32[(long)(k0 + k + 1) * N2 + (n0 >> 1) + n2];
            int n = 2 * n2;
            Bt[n][kp]     = (a & 0xFFFFu) | (b << 16);
            Bt[n + 1][kp] = (a >> 16)     | (b & 0xFFFF0000u);
        }
        __syncthreads();

        // ---- A fragments: 2 x ds_load_b128 each
        v8u au[2];
        #pragma unroll
        for (int mt = 0; mt < 2; ++mt) {
            const uint4* rowp = (const uint4*)&As[wm + mt * 16 + lm][0];
            uint4 lo = rowp[hiL ? 1 : 0];
            uint4 hi = rowp[(hiL ? 1 : 0) + 2];
            au[mt][0] = lo.x; au[mt][1] = lo.y; au[mt][2] = lo.z; au[mt][3] = lo.w;
            au[mt][4] = hi.x; au[mt][5] = hi.y; au[mt][6] = hi.z; au[mt][7] = hi.w;
        }
        // ---- B fragments: 2 x ds_load_b128 each
        v8u bu[4];
        #pragma unroll
        for (int nt = 0; nt < 4; ++nt) {
            const uint4* colp = (const uint4*)&Bt[wn + nt * 16 + lm][0];
            uint4 lo = colp[hiL ? 1 : 0];
            uint4 hi = colp[(hiL ? 1 : 0) + 2];
            bu[nt][0] = lo.x; bu[nt][1] = lo.y; bu[nt][2] = lo.z; bu[nt][3] = lo.w;
            bu[nt][4] = hi.x; bu[nt][5] = hi.y; bu[nt][6] = hi.z; bu[nt][7] = hi.w;
        }
        #pragma unroll
        for (int mt = 0; mt < 2; ++mt)
            #pragma unroll
            for (int nt = 0; nt < 4; ++nt)
                acc[mt][nt] = __builtin_amdgcn_wmma_f32_16x16x32_bf16(
                    false, asbf(au[mt]), false, asbf(bu[nt]),
                    (short)0, acc[mt][nt], false, false);
    }

    #pragma unroll
    for (int mt = 0; mt < 2; ++mt)
        #pragma unroll
        for (int nt = 0; nt < 4; ++nt)
            #pragma unroll
            for (int r = 0; r < 8; ++r) {
                long row = m0 + wm + mt * 16 + r + (hiL ? 8 : 0);
                long col = n0 + wn + nt * 16 + lm;
                float v = acc[mt][nt][r];
                if constexpr (BF16OUT) Cb[row * N + col] = f2bf(v);
                else                   Cf[row * N + col] = v;
            }
}

// ---------------------------------------------------------------------------
// Fused causal flash attention (bf16 WMMA, f32 online softmax).
// A block's 8 waves process 8 consecutive 16-query tiles of ONE (b,h), so
// K/V tiles are staged to LDS cooperatively once per 32-key block.
// Computes S^T = K*Q^T so softmax is per-lane and P^T(D) repacks in-lane to
// P(A-layout). All fragment reads are ds_load_b128.
// qkv: [B,T,3C] bf16 (q|k|v, C=1024=16 heads*64). yout: [B,T,C] bf16.
// ---------------------------------------------------------------------------
__global__ __launch_bounds__(256) void attn_flash_wmma(
    const unsigned short* __restrict__ qkv,
    unsigned short* __restrict__ yout)
{
    const int T = 2048;
    __shared__ __align__(16) unsigned Ks[32][36];  // [key][d/2] pairs (32 used)
    __shared__ __align__(16) unsigned Vt[64][20];  // [d][key-pair]  (16 used)

    const int t    = threadIdx.x;
    const int wid  = t >> 5;
    const int lane = t & 31;
    const int lm   = lane & 15;
    const bool hiL = lane >= 16;

    // block-uniform (b,h); 8 consecutive query tiles per block
    const int Wb  = blockIdx.x * 8;
    const int qtb = Wb & 127;            // first query tile of block
    const int h   = (Wb >> 7) & 15;
    const int b   = Wb >> 11;
    const int t0  = (qtb + wid) * 16;    // this wave's first query row

    const unsigned* qkv32 = (const unsigned*)qkv;   // row stride 1536 u32
    const long rowK = (long)(b * T) * 1536 + 512 + h * 32;  // K base (u32)
    const long rowV = (long)(b * T) * 1536 + 1024 + h * 32; // V base (u32)

    // ---- Q as B-fragments (Q^T), 2 chunks of d (0..31, 32..63)
    v8u bq[2];
    {
        long qb = (long)(b * T + (t0 + lm)) * 1536 + h * 32;
        #pragma unroll
        for (int c = 0; c < 2; ++c)
            #pragma unroll
            for (int v = 0; v < 8; ++v) {
                int d = 32 * c + (hiL ? 16 : 0) + 2 * v;
                bq[c][v] = qkv32[qb + (d >> 1)];
            }
    }

    v8f accy[4] = {};
    float m_run = -3.0e38f, l_run = 0.0f;
    const int jmax_own = (t0 + 15) >> 5;
    const int jmax_blk = ((qtb + 7) * 16 + 15) >> 5;   // max over block's waves

    for (int jj = 0; jj <= jmax_blk; ++jj) {
        const int kt0 = jj << 5;
        __syncthreads();
        // ---- stage K tile (32 keys x 64 d): one b128 load+store per thread
        {
            int r = t >> 3;              // key row 0..31
            int g = t & 7;               // uint4 group 0..7 (8 * 4 dwords = 32)
            const uint4* src = (const uint4*)&qkv32[rowK + (long)(kt0 + r) * 1536];
            ((uint4*)&Ks[r][0])[g] = src[g];
        }
        // ---- stage V tile transposed: coalesced u32 -> key-pair pack -> LDS
        #pragma unroll
        for (int i = 0; i < 2; ++i) {
            int flat = i * 256 + t;
            int d2 = flat & 31;          // d pair index 0..31
            int kp = flat >> 5;          // key pair index 0..15
            long base = rowV + (long)(kt0 + 2 * kp) * 1536 + d2;
            unsigned a  = qkv32[base];
            unsigned bb = qkv32[base + 1536];
            int d = 2 * d2;
            Vt[d][kp]     = (a & 0xFFFFu) | (bb << 16);
            Vt[d + 1][kp] = (a >> 16)     | (bb & 0xFFFF0000u);
        }
        __syncthreads();

        if (jj <= jmax_own) {            // wave-uniform causal skip
            float S[2][8];
            // ---- S^T for two 16-key subtiles: A = K-tile (from LDS), B = Q^T
            #pragma unroll
            for (int s = 0; s < 2; ++s) {
                const uint4* krow = (const uint4*)&Ks[16 * s + lm][0];
                v8f sacc;
                #pragma unroll
                for (int c = 0; c < 2; ++c) {
                    int g = 4 * c + (hiL ? 2 : 0);
                    uint4 lo = krow[g];
                    uint4 hi = krow[g + 1];
                    v8u ak;
                    ak[0] = lo.x; ak[1] = lo.y; ak[2] = lo.z; ak[3] = lo.w;
                    ak[4] = hi.x; ak[5] = hi.y; ak[6] = hi.z; ak[7] = hi.w;
                    if (c == 0) {
                        v8f z = {};
                        sacc = __builtin_amdgcn_wmma_f32_16x16x32_bf16(
                            false, asbf(ak), false, asbf(bq[0]), (short)0, z, false, false);
                    } else {
                        sacc = __builtin_amdgcn_wmma_f32_16x16x32_bf16(
                            false, asbf(ak), false, asbf(bq[1]), (short)0, sacc, false, false);
                    }
                }
                int q_t = t0 + lm;
                #pragma unroll
                for (int r = 0; r < 8; ++r) {
                    int key = kt0 + 16 * s + r + (hiL ? 8 : 0);
                    float sv = sacc[r] * 0.125f;           // 1/sqrt(64)
                    S[s][r] = (key > q_t) ? -1.0e30f : sv; // causal mask
                }
            }
            // ---- online softmax (per-lane = per-query; merge halves xor16)
            float mt = S[0][0];
            #pragma unroll
            for (int s = 0; s < 2; ++s)
                #pragma unroll
                for (int r = 0; r < 8; ++r) mt = fmaxf(mt, S[s][r]);
            mt = fmaxf(mt, __shfl_xor(mt, 16, 32));
            float new_m = fmaxf(m_run, mt);
            float alpha = __expf(m_run - new_m);
            float p[2][8], ladd = 0.0f;
            #pragma unroll
            for (int s = 0; s < 2; ++s)
                #pragma unroll
                for (int r = 0; r < 8; ++r) {
                    p[s][r] = __expf(S[s][r] - new_m);
                    ladd += p[s][r];
                }
            ladd += __shfl_xor(ladd, 16, 32);
            l_run = l_run * alpha + ladd;
            m_run = new_m;

            // ---- P^T(D-layout) -> P(A-layout): pure in-lane packing
            v8u ap;
            #pragma unroll
            for (int v = 0; v < 4; ++v) ap[v]     = packbf(p[0][2 * v], p[0][2 * v + 1]);
            #pragma unroll
            for (int v = 0; v < 4; ++v) ap[4 + v] = packbf(p[1][2 * v], p[1][2 * v + 1]);

            // ---- broadcast alpha to y D-layout rows
            float al[8];
            #pragma unroll
            for (int r = 0; r < 8; ++r) al[r] = __shfl(alpha, r + (hiL ? 8 : 0), 32);

            // ---- y += P * V ; V B-fragments from Vt, 2 x ds_load_b128 each
            #pragma unroll
            for (int dt = 0; dt < 4; ++dt) {
                const uint4* colp = (const uint4*)&Vt[dt * 16 + lm][0];
                uint4 lo = colp[hiL ? 1 : 0];
                uint4 hi = colp[(hiL ? 1 : 0) + 2];
                v8u bv;
                bv[0] = lo.x; bv[1] = lo.y; bv[2] = lo.z; bv[3] = lo.w;
                bv[4] = hi.x; bv[5] = hi.y; bv[6] = hi.z; bv[7] = hi.w;
                #pragma unroll
                for (int r = 0; r < 8; ++r) accy[dt][r] *= al[r];
                accy[dt] = __builtin_amdgcn_wmma_f32_16x16x32_bf16(
                    false, asbf(ap), false, asbf(bv), (short)0, accy[dt], false, false);
            }
        }
    }

    // ---- normalize by per-query l and store bf16
    float li[8];
    #pragma unroll
    for (int r = 0; r < 8; ++r)
        li[r] = 1.0f / __shfl(l_run, r + (hiL ? 8 : 0), 32);
    #pragma unroll
    for (int dt = 0; dt < 4; ++dt)
        #pragma unroll
        for (int r = 0; r < 8; ++r) {
            long row = t0 + r + (hiL ? 8 : 0);
            long col = h * 64 + dt * 16 + lm;
            yout[(long)(b * T + row) * 1024 + col] = f2bf(accy[dt][r] * li[r]);
        }
}

// ---------------------------------------------------------------------------
// Launcher
// ---------------------------------------------------------------------------
extern "C" void kernel_launch(void* const* d_in, const int* in_sizes, int n_in,
                              void* d_out, int out_size, void* d_ws, size_t ws_size,
                              hipStream_t stream) {
    const int B = 4, T = 2048, C = 1024;
    const int M = B * T;                 // 8192
    const float* x  = (const float*)d_in[0];   // [B,T,C]
    const float* Wa = (const float*)d_in[1];   // [C,3C]
    const float* Wp = (const float*)d_in[2];   // [C,C]
    float* out = (float*)d_out;                // [B,T,C] f32

    // workspace layout (bf16 = ushort); all offsets 16B aligned
    unsigned short* x_bf   = (unsigned short*)d_ws;            // M*C
    unsigned short* wa_bf  = x_bf  + (size_t)M * C;            // C*3C
    unsigned short* wp_bf  = wa_bf + (size_t)C * 3 * C;        // C*C
    unsigned short* qkv_bf = wp_bf + (size_t)C * C;            // M*3C
    unsigned short* y_bf   = qkv_bf + (size_t)M * 3 * C;       // M*C

    // 1) fp32 -> bf16 converts
    cvt_f32_bf16<<<4096, 256, 0, stream>>>(x,  x_bf,  M * C);
    cvt_f32_bf16<<<4096, 256, 0, stream>>>(Wa, wa_bf, C * 3 * C);
    cvt_f32_bf16<<<2048, 256, 0, stream>>>(Wp, wp_bf, C * C);

    // 2) qkv = x @ W_attn   (M x 3C, bf16 out)
    wmma_gemm_bf16<true><<<dim3(3 * C / 128, M / 128), 256, 0, stream>>>(
        x_bf, wa_bf, qkv_bf, nullptr, M, 3 * C, C);

    // 3) flash attention: B*H*(T/16) = 8192 waves = 1024 blocks x 8 waves
    attn_flash_wmma<<<dim3(1024), 256, 0, stream>>>(qkv_bf, y_bf);

    // 4) out = y @ W_proj   (M x C, f32 out)
    wmma_gemm_bf16<false><<<dim3(C / 128, M / 128), 256, 0, stream>>>(
        y_bf, wp_bf, nullptr, out, M, C, C);
}
